// MaskedScaleDotProductAttention_12695923327537
// MI455X (gfx1250) — matrix operand order repaired and checked by reading, not compiled, not run on hardware
//
#include <hip/hip_runtime.h>
#include <cstdint>

// ---------------------------------------------------------------------------
// MI455X / gfx1250 fused attention block, bf16 WMMA pipeline:
//   pre:   q,k,Wq,Wk,Wfc -> bf16; v -> vT[b,d,t] bf16 (LDS-tiled transpose)
//   qp = qbf@Wq^T * (1/sqrt(512)); kp = kbf@Wk^T      (all-bf16 WMMA)
//   scores = qp @ kp^T -> softmax -> attn (output #2, fp32)
//   out1 = attn @ v + v                               (bf16 WMMA, fp32 acc)
//   out  = LayerNorm(out1 @ Wfc^T)                    (output #1, fp32)
// score/av GEMMs stage the shared B tile in LDS via double-buffered
// global_load_async_to_lds_b128 (ASYNCcnt), amortized across 8 waves.
// ---------------------------------------------------------------------------

constexpr int BB = 8;
constexpr int SS = 2048;
constexpr int DD = 512;
constexpr int DKK = 512;

// LDS B-tile: 64 rows x 32 bf16, padded to 40 ushorts (80 B) per row.
constexpr int BPITCH = 40;                    // ushorts per row
constexpr unsigned BUFB = 64 * BPITCH * 2;    // bytes per buffer (5120)

typedef __attribute__((ext_vector_type(16))) __bf16 v16bf;
typedef __attribute__((ext_vector_type(8)))  float  v8f;
typedef __attribute__((ext_vector_type(16))) unsigned short v16us;

__device__ __forceinline__ unsigned short f2bf_bits(float f) {
  return __builtin_bit_cast(unsigned short, (__bf16)f);  // native cvt, RTNE
}

__device__ __forceinline__ v8f wmma_bf16(v16bf a, v16bf b, v8f c) {
  return __builtin_amdgcn_wmma_f32_16x16x32_bf16(false, a, false, b, (short)0, c,
                                                 false, false);
}

// Async 16-byte global -> LDS copy (per-lane addresses), tracked by ASYNCcnt.
__device__ __forceinline__ void async_copy16(unsigned lds_off, const void* gptr) {
  asm volatile("global_load_async_to_lds_b128 %0, %1, off"
               :: "v"(lds_off), "v"((unsigned long long)(uintptr_t)gptr)
               : "memory");
}
__device__ __forceinline__ void wait_async0() {
  asm volatile("s_wait_asynccnt 0x0" ::: "memory");
}

// ---- A fragment: 16x32 (MxK) 16-bit. lane r=tid&15 -> row M=r, h=tid>>4.
//      element e -> K = (e>>3)*16 + h*8 + (e&7)   (two contiguous 8-runs)
__device__ __forceinline__ v16bf fragA_bf16(const unsigned short* __restrict__ p, int h) {
  v16us u;
  const unsigned short* p0 = p + h * 8;
  const unsigned short* p1 = p + 16 + h * 8;
#pragma unroll
  for (int j = 0; j < 8; ++j) u[j] = p0[j];
#pragma unroll
  for (int j = 0; j < 8; ++j) u[8 + j] = p1[j];
  return __builtin_bit_cast(v16bf, u);
}

__device__ __forceinline__ v16bf fragA_f32_cvt(const float* __restrict__ p, int h) {
  v16bf u;
  const float* p0 = p + h * 8;
  const float* p1 = p + 16 + h * 8;
#pragma unroll
  for (int j = 0; j < 8; ++j) u[j] = (__bf16)p0[j];
#pragma unroll
  for (int j = 0; j < 8; ++j) u[8 + j] = (__bf16)p1[j];
  return u;
}

// ---- B fragment: 32x16 (KxN) 16-bit. lane r -> col N=r, h=tid>>4.
//      element e -> K = (e>>3)*16 + h*8 + (e&7)   (two contiguous 8-runs)
__device__ __forceinline__ v16bf fragB_bf16(const unsigned short* p, int h) {
  v16us u;
  const unsigned short* p0 = p + h * 8;
#pragma unroll
  for (int e = 0; e < 16; ++e) u[e] = p0[(e >> 3) * 16 + (e & 7)];
  return __builtin_bit_cast(v16bf, u);
}

// ===========================================================================
// Pre-pass: elementwise fp32 -> bf16
// ===========================================================================
__global__ __launch_bounds__(256) void cvt_f32_bf16(const float* __restrict__ in,
                                                    unsigned short* __restrict__ outp) {
  const size_t i = ((size_t)blockIdx.x * 256 + threadIdx.x) * 4;
  const float4 f = *(const float4*)(in + i);
  ushort4 o;
  o.x = f2bf_bits(f.x);
  o.y = f2bf_bits(f.y);
  o.z = f2bf_bits(f.z);
  o.w = f2bf_bits(f.w);
  *(ushort4*)(outp + i) = o;
}

// ===========================================================================
// Pre-pass: v[b,t,d] fp32 -> vT[b,d,t] bf16, 32x32 LDS tiles
// ===========================================================================
__global__ __launch_bounds__(256) void transpose_cvt_v(const float* __restrict__ V,
                                                       unsigned short* __restrict__ vT) {
  __shared__ float tile[32][33];
  const int b  = blockIdx.z;
  const int t0 = blockIdx.x * 32;
  const int d0 = blockIdx.y * 32;
  const int tx  = threadIdx.x & 31;
  const int ty4 = threadIdx.x >> 5;
#pragma unroll
  for (int j = 0; j < 4; ++j) {
    const int ty = ty4 * 4 + j;
    tile[ty][tx] = V[(size_t)(b * SS + t0 + ty) * DD + d0 + tx];
  }
  __syncthreads();
#pragma unroll
  for (int j = 0; j < 4; ++j) {
    const int dy = ty4 * 4 + j;
    vT[((size_t)b * DD + d0 + dy) * SS + t0 + tx] = f2bf_bits(tile[tx][dy]);
  }
}

// ===========================================================================
// Kernel: Y[m,n] = alpha * sum_k X[m,k]*W[n,k], all bf16 (W is L2-resident).
// 8 waves/block; block tile 128(M) x 64(N); wave tile 16 x 64.
// ===========================================================================
__global__ __launch_bounds__(256) void proj_gemm(const unsigned short* __restrict__ X,
                                                 const unsigned short* __restrict__ W,
                                                 unsigned short* __restrict__ Y,
                                                 float alpha) {
  const int lane = threadIdx.x & 31;
  const int wid  = threadIdx.x >> 5;
  const int r = lane & 15, h = lane >> 4;
  const int n0 = blockIdx.x * 64;
  const int m0 = blockIdx.y * 128 + wid * 16;

  v8f acc[4] = {};
  const unsigned short* xrow = X + (size_t)(m0 + r) * DD;
  for (int k0 = 0; k0 < DD; k0 += 32) {
    v16bf a = fragA_bf16(xrow + k0, h);
#pragma unroll
    for (int nt = 0; nt < 4; ++nt) {
      v16bf b = fragB_bf16(W + (size_t)(n0 + nt * 16 + r) * DD + k0, h);
      acc[nt] = wmma_bf16(a, b, acc[nt]);
    }
  }
#pragma unroll
  for (int nt = 0; nt < 4; ++nt) {
    const int col = n0 + nt * 16 + r;
#pragma unroll
    for (int g = 0; g < 8; ++g) {
      const int row = m0 + g + 8 * h;
      Y[(size_t)row * DKK + col] = f2bf_bits(acc[nt][g] * alpha);
    }
  }
}

// ===========================================================================
// Kernel: scores[b,q,t] = sum_k qp[b,q,k]*kp[b,t,k]  (bf16 in, fp32 out)
// Wave tile 32(M) x 64(N); B tile (64x32 bf16) async-staged in LDS,
// double-buffered, shared by all 8 waves.
// ===========================================================================
__global__ __launch_bounds__(256) void score_gemm(const unsigned short* __restrict__ qp,
                                                  const unsigned short* __restrict__ kp,
                                                  float* __restrict__ attn) {
  __shared__ unsigned short bsm[2][64 * BPITCH];  // 2 x 5 KB
  const int t = threadIdx.x;
  const int lane = t & 31, wid = t >> 5;
  const int r = lane & 15, h = lane >> 4;
  const int b  = blockIdx.z;
  const int n0 = blockIdx.x * 64;
  const int m0 = blockIdx.y * 256 + wid * 32;

  // this thread's async-copy slot: 16 B chunk (row, chunk) of the B tile
  const int crow = t >> 2, cchunk = t & 3;
  const unsigned short* gsrc = kp + (size_t)(b * SS + n0 + crow) * DKK + cchunk * 8;
  const unsigned lds_my0 =
      (unsigned)(uintptr_t)(&bsm[0][0]) + (unsigned)(crow * BPITCH + cchunk * 8) * 2u;

  v8f acc[8] = {};
  const unsigned short* arow0 = qp + (size_t)(b * SS + m0 + r) * DKK;
  const unsigned short* arow1 = arow0 + (size_t)16 * DKK;

  async_copy16(lds_my0, gsrc);  // prologue: K tile 0 -> buf0
  constexpr int NK = DKK / 32;  // 16
  for (int kb = 0; kb < NK; ++kb) {
    const int cur = kb & 1;
    wait_async0();              // my chunk of buf[cur] landed
    __syncthreads();            // everyone's chunks landed; prev reads done
    if (kb + 1 < NK)            // stream next K tile into the other buffer
      async_copy16(lds_my0 + (cur ? 0u : BUFB), gsrc + (kb + 1) * 32);

    const int k0 = kb * 32;
    __builtin_prefetch(arow0 + k0 + 128, 0, 1);
    v16bf a0 = fragA_bf16(arow0 + k0, h);
    v16bf a1 = fragA_bf16(arow1 + k0, h);
#pragma unroll
    for (int nt = 0; nt < 4; ++nt) {
      v16bf bf = fragB_bf16(&bsm[cur][(nt * 16 + r) * BPITCH], h);
      acc[nt]     = wmma_bf16(a0, bf, acc[nt]);
      acc[4 + nt] = wmma_bf16(a1, bf, acc[4 + nt]);
    }
  }

  float* abase = attn + (size_t)b * SS * SS;
#pragma unroll
  for (int half = 0; half < 2; ++half) {
#pragma unroll
    for (int nt = 0; nt < 4; ++nt) {
      const int col = n0 + nt * 16 + r;
#pragma unroll
      for (int g = 0; g < 8; ++g) {
        const int row = m0 + half * 16 + g + 8 * h;
        abase[(size_t)row * SS + col] = acc[half * 4 + nt][g];
      }
    }
  }
}

// ===========================================================================
// Kernel: row softmax over 2048 elements, in place (* GAMMA == 1.0)
// ===========================================================================
__global__ __launch_bounds__(256) void softmax_rows(float* __restrict__ attn) {
  float* p = attn + (size_t)blockIdx.x * SS;
  const int t = threadIdx.x;
  const int lane = t & 31, wid = t >> 5;
  __shared__ float redm[8];
  __shared__ float reds[8];

  float vals[8];
  float m = -3.4e38f;
#pragma unroll
  for (int j = 0; j < 8; ++j) {
    vals[j] = p[t + 256 * j];
    m = fmaxf(m, vals[j]);
  }
#pragma unroll
  for (int off = 16; off > 0; off >>= 1) m = fmaxf(m, __shfl_xor(m, off, 32));
  if (lane == 0) redm[wid] = m;
  __syncthreads();
  float gm = redm[0];
#pragma unroll
  for (int i = 1; i < 8; ++i) gm = fmaxf(gm, redm[i]);

  float s = 0.f;
#pragma unroll
  for (int j = 0; j < 8; ++j) {
    vals[j] = __expf(vals[j] - gm);
    s += vals[j];
  }
#pragma unroll
  for (int off = 16; off > 0; off >>= 1) s += __shfl_xor(s, off, 32);
  if (lane == 0) reds[wid] = s;
  __syncthreads();
  float gs = 0.f;
#pragma unroll
  for (int i = 0; i < 8; ++i) gs += reds[i];
  const float inv = 1.0f / gs;
#pragma unroll
  for (int j = 0; j < 8; ++j) p[t + 256 * j] = vals[j] * inv;
}

// ===========================================================================
// Kernel: out1[b,q,d] = sum_t attn[b,q,t]*v[b,t,d] + v[b,q,d]  (bf16 out)
// A: attn fp32 -> bf16 in-register; B: vT[b,d,t] async-staged in LDS.
// Wave tile 32(M) x 64(N).
// ===========================================================================
__global__ __launch_bounds__(256) void av_gemm(const float* __restrict__ attn,
                                               const unsigned short* __restrict__ vT,
                                               const float* __restrict__ V,
                                               unsigned short* __restrict__ out1) {
  __shared__ unsigned short bsm[2][64 * BPITCH];
  const int t = threadIdx.x;
  const int lane = t & 31, wid = t >> 5;
  const int r = lane & 15, h = lane >> 4;
  const int b  = blockIdx.z;
  const int n0 = blockIdx.x * 64;
  const int m0 = blockIdx.y * 256 + wid * 32;

  const int crow = t >> 2, cchunk = t & 3;
  const unsigned short* gsrc = vT + ((size_t)b * DD + n0 + crow) * SS + cchunk * 8;
  const unsigned lds_my0 =
      (unsigned)(uintptr_t)(&bsm[0][0]) + (unsigned)(crow * BPITCH + cchunk * 8) * 2u;

  v8f acc[8] = {};
  const float* arow0 = attn + (size_t)b * SS * SS + (size_t)(m0 + r) * SS;
  const float* arow1 = arow0 + (size_t)16 * SS;

  async_copy16(lds_my0, gsrc);
  constexpr int NK = SS / 32;  // 64
  for (int kb = 0; kb < NK; ++kb) {
    const int cur = kb & 1;
    wait_async0();
    __syncthreads();
    if (kb + 1 < NK)
      async_copy16(lds_my0 + (cur ? 0u : BUFB), gsrc + (kb + 1) * 32);

    const int k0 = kb * 32;
    __builtin_prefetch(arow0 + k0 + 128, 0, 1);
    v16bf a0 = fragA_f32_cvt(arow0 + k0, h);
    v16bf a1 = fragA_f32_cvt(arow1 + k0, h);
#pragma unroll
    for (int nt = 0; nt < 4; ++nt) {
      v16bf bf = fragB_bf16(&bsm[cur][(nt * 16 + r) * BPITCH], h);
      acc[nt]     = wmma_bf16(a0, bf, acc[nt]);
      acc[4 + nt] = wmma_bf16(a1, bf, acc[4 + nt]);
    }
  }
#pragma unroll
  for (int half = 0; half < 2; ++half) {
#pragma unroll
    for (int nt = 0; nt < 4; ++nt) {
      const int col = n0 + nt * 16 + r;
#pragma unroll
      for (int g = 0; g < 8; ++g) {
        const int row = m0 + half * 16 + g + 8 * h;
        const size_t idx = (size_t)(b * SS + row) * DD + col;
        out1[idx] = f2bf_bits(acc[half * 4 + nt][g] + V[idx]);  // residual
      }
    }
  }
}

// ===========================================================================
// Kernel: out[m,n] = LN( sum_k out1[m,k]*Wfc[n,k] ), fp32 out.
// Block owns full 16(M) x 512(N) strip; LDS-staged fused LayerNorm.
// ===========================================================================
__global__ __launch_bounds__(256) void fc_ln(const unsigned short* __restrict__ out1,
                                             const unsigned short* __restrict__ Wfc,
                                             const float* __restrict__ lnw,
                                             const float* __restrict__ lnb,
                                             float* __restrict__ out) {
  __shared__ float tile[16 * 512];  // 32 KB
  const int lane = threadIdx.x & 31;
  const int wid  = threadIdx.x >> 5;
  const int r = lane & 15, h = lane >> 4;
  const int row0 = blockIdx.x * 16;
  const int n0 = wid * 64;

  v8f acc[4] = {};
  const unsigned short* arow = out1 + (size_t)(row0 + r) * DD;
  for (int k0 = 0; k0 < DD; k0 += 32) {
    v16bf a = fragA_bf16(arow + k0, h);
#pragma unroll
    for (int nt = 0; nt < 4; ++nt) {
      v16bf b = fragB_bf16(Wfc + (size_t)(n0 + nt * 16 + r) * DD + k0, h);
      acc[nt] = wmma_bf16(a, b, acc[nt]);
    }
  }
#pragma unroll
  for (int nt = 0; nt < 4; ++nt) {
    const int col = n0 + nt * 16 + r;
#pragma unroll
    for (int g = 0; g < 8; ++g) tile[(g + 8 * h) * 512 + col] = acc[nt][g];
  }
  __syncthreads();

  const int row = threadIdx.x >> 4;
  const int i   = threadIdx.x & 15;
  const float* trow = tile + row * 512;
  float xs[32];
  float sum = 0.f, sumsq = 0.f;
#pragma unroll
  for (int j = 0; j < 32; ++j) {
    const float x = trow[i * 32 + j];
    xs[j] = x;
    sum += x;
    sumsq += x * x;
  }
#pragma unroll
  for (int off = 8; off > 0; off >>= 1) {
    sum   += __shfl_xor(sum, off, 16);
    sumsq += __shfl_xor(sumsq, off, 16);
  }
  const float mean = sum * (1.0f / 512.0f);
  const float var  = sumsq * (1.0f / 512.0f) - mean * mean;
  const float rs   = rsqrtf(var + 1e-6f);
  float* orow = out + (size_t)(row0 + row) * DD;
#pragma unroll
  for (int j = 0; j < 32; ++j) {
    const int c = i * 32 + j;
    orow[c] = (xs[j] - mean) * rs * lnw[c] + lnb[c];
  }
}

// ===========================================================================
extern "C" void kernel_launch(void* const* d_in, const int* in_sizes, int n_in,
                              void* d_out, int out_size, void* d_ws, size_t ws_size,
                              hipStream_t stream) {
  const float* q    = (const float*)d_in[0];
  const float* k    = (const float*)d_in[1];
  const float* v    = (const float*)d_in[2];
  const float* w_qs = (const float*)d_in[3];
  const float* w_ks = (const float*)d_in[4];
  const float* w_fc = (const float*)d_in[5];
  const float* ln_w = (const float*)d_in[6];
  const float* ln_b = (const float*)d_in[7];

  float* out  = (float*)d_out;                       // [B,S,D]
  float* attn = out + (size_t)BB * SS * DD;          // [B,S,S] (output #2)

  char* ws = (char*)d_ws;
  const size_t act = (size_t)BB * SS * DKK * sizeof(unsigned short);  // 16 MiB
  const size_t wsz = (size_t)DKK * DD * sizeof(unsigned short);       // 512 KiB
  unsigned short* qbf   = (unsigned short*)(ws + 0 * act);
  unsigned short* kbf   = (unsigned short*)(ws + 1 * act);
  unsigned short* vT    = (unsigned short*)(ws + 2 * act);
  unsigned short* qp    = (unsigned short*)(ws + 3 * act);
  unsigned short* kp    = (unsigned short*)(ws + 4 * act);
  unsigned short* wq_bf = (unsigned short*)(ws + 5 * act);
  unsigned short* wk_bf = (unsigned short*)(ws + 5 * act + wsz);
  unsigned short* wf_bf = (unsigned short*)(ws + 5 * act + 2 * wsz);
  unsigned short* o1    = qbf;  // alias: qbf dead after proj_gemm(q)

  const dim3 blk(256);
  const float inv_scale = 0.04419417382415922f;  // 1/sqrt(512)
  const int n_act = BB * SS * DD;
  const int n_w   = DKK * DD;

  cvt_f32_bf16<<<dim3(n_act / 1024), blk, 0, stream>>>(q, qbf);
  cvt_f32_bf16<<<dim3(n_act / 1024), blk, 0, stream>>>(k, kbf);
  cvt_f32_bf16<<<dim3(n_w / 1024),  blk, 0, stream>>>(w_qs, wq_bf);
  cvt_f32_bf16<<<dim3(n_w / 1024),  blk, 0, stream>>>(w_ks, wk_bf);
  cvt_f32_bf16<<<dim3(n_w / 1024),  blk, 0, stream>>>(w_fc, wf_bf);
  transpose_cvt_v<<<dim3(SS / 32, DD / 32, BB), blk, 0, stream>>>(v, vT);

  proj_gemm<<<dim3(DKK / 64, (BB * SS) / 128), blk, 0, stream>>>(qbf, wq_bf, qp, inv_scale);
  proj_gemm<<<dim3(DKK / 64, (BB * SS) / 128), blk, 0, stream>>>(kbf, wk_bf, kp, 1.0f);
  score_gemm<<<dim3(SS / 64, SS / 256, BB), blk, 0, stream>>>(qp, kp, attn);
  softmax_rows<<<dim3(BB * SS), blk, 0, stream>>>(attn);
  av_gemm<<<dim3(DD / 64, SS / 256, BB), blk, 0, stream>>>(attn, vT, v, o1);
  fc_ln<<<dim3((BB * SS) / 16), blk, 0, stream>>>(o1, wf_bf, ln_w, ln_b, out);
}
